// NestRQModel_5823975653922
// MI455X (gfx1250) — matrix operand order repaired and checked by reading, not compiled, not run on hardware
//
#include <hip/hip_runtime.h>
#include <math.h>

// ---- problem constants ----
#define BATCH   16
#define TENC    512          // (2048-4)/4+1
#define ENCD    512
#define NEMB    8192
#define EMBD    16
#define INDIM   320          // 80*4
#define NSLICE  4            // N split per row-tile (2048 cols each)

typedef __attribute__((ext_vector_type(16))) __bf16 v16bf;
typedef __attribute__((ext_vector_type(8)))  float  v8f;

union Frag { unsigned int u[8]; uint4 q[2]; v16bf v; };

static __device__ __forceinline__ unsigned short f2bf(float f) {
    unsigned int u = __float_as_uint(f);
    u += 0x7FFFu + ((u >> 16) & 1u);       // round-to-nearest-even
    return (unsigned short)(u >> 16);
}

// LDS byte offset of a __shared__ object (generic -> addrspace(3) -> 32-bit offset)
typedef __attribute__((address_space(3))) const void* lds_cptr_t;
static __device__ __forceinline__ unsigned lds_off(const void* p) {
    return (unsigned)(unsigned long long)(lds_cptr_t)p;
}

// one lane copies 16 bytes global -> LDS via the async DMA path (ASYNCcnt)
static __device__ __forceinline__ void async_cp16(unsigned ldsAddr, const void* gptr) {
    unsigned long long ga = (unsigned long long)gptr;
    asm volatile("global_load_async_to_lds_b128 %0, %1, off"
                 :: "v"(ldsAddr), "v"(ga) : "memory");
}
static __device__ __forceinline__ void wait_async0() {
    asm volatile("s_wait_asynccnt 0x0" ::: "memory");
}

// ---------------- init: zero presence bitmap + 3 accumulators ----------------
__global__ void k_init(int* __restrict__ presence, float* __restrict__ acc3) {
    int i = blockIdx.x * blockDim.x + threadIdx.x;
    if (i < NEMB) presence[i] = 0;
    if (i < 3)    acc3[i] = 0.f;
}

// ---- pack top_n_out (512 x 8192 f32) into WMMA-B fragment-tiled bf16 layout ----
// Wp2 flat index: (((nt*16 + kk)*32 + lane)*8 + v)  where
//   column n = nt*16 + (lane&15),  K = kk*32 + (lane>>4)*16 + 2*v  (+pair k+1 in hi16)
__global__ void k_packW(const float* __restrict__ W, unsigned int* __restrict__ Wp2) {
    int id   = blockIdx.x * 256 + threadIdx.x;     // 2,097,152
    int v    = id & 7;
    int lane = (id >> 3) & 31;
    int kk   = (id >> 8) & 15;
    int nt   = id >> 12;
    int n    = nt * 16 + (lane & 15);
    int k    = kk * 32 + (lane >> 4) * 16 + 2 * v;
    float a = W[(size_t)k * NEMB + n];
    float b = W[(size_t)(k + 1) * NEMB + n];
    Wp2[id] = (unsigned)f2bf(a) | ((unsigned)f2bf(b) << 16);
}

// ---- pack encoder_out (8192 rows x 512 f32) -> bf16 K-pairs: Apk[row*256+k2] ----
__global__ void k_packA(const float* __restrict__ E, unsigned int* __restrict__ Apk) {
    int id  = blockIdx.x * 256 + threadIdx.x;      // 8192*256
    int k2  = id & 255;
    int row = id >> 8;
    const float* p = E + (size_t)row * ENCD + 2 * k2;
    Apk[id] = (unsigned)f2bf(p[0]) | ((unsigned)f2bf(p[1]) << 16);
}

// ---- stack+LN+project+normalize+nearest-embedding : one wave per (b,n) row ----
__global__ __launch_bounds__(256) void k_codes(const float* __restrict__ feats,
                                               const int* __restrict__ lens,
                                               const float* __restrict__ proj,
                                               const float* __restrict__ emb,
                                               int* __restrict__ codes,
                                               int* __restrict__ smask) {
    int wave = threadIdx.x >> 5, lane = threadIdx.x & 31;
    int row  = blockIdx.x * 8 + wave;              // 8192 rows
    int b    = row >> 9, n = row & 511;

    float s[10]; float sum = 0.f, sq = 0.f;
    #pragma unroll
    for (int i = 0; i < 10; ++i) {
        int j = lane * 10 + i;
        int frame = j / 80, mel = j % 80;
        float v = feats[((size_t)b * 2048 + (size_t)n * 4 + frame) * 80 + mel];
        s[i] = v; sum += v; sq += v * v;
    }
    for (int off = 16; off > 0; off >>= 1) { sum += __shfl_xor(sum, off); sq += __shfl_xor(sq, off); }
    float mean = sum * (1.f / INDIM);
    float var  = sq * (1.f / INDIM) - mean * mean;
    float rs   = rsqrtf(var + 1e-6f);

    float acc[EMBD];
    #pragma unroll
    for (int c = 0; c < EMBD; ++c) acc[c] = 0.f;
    #pragma unroll
    for (int i = 0; i < 10; ++i) {
        float v = (s[i] - mean) * rs;
        const float* pr = proj + (size_t)(lane * 10 + i) * EMBD;
        #pragma unroll
        for (int c = 0; c < EMBD; ++c) acc[c] += v * pr[c];
    }
    #pragma unroll
    for (int c = 0; c < EMBD; ++c)
        for (int off = 16; off > 0; off >>= 1) acc[c] += __shfl_xor(acc[c], off);

    float nrm = 0.f;
    #pragma unroll
    for (int c = 0; c < EMBD; ++c) nrm += acc[c] * acc[c];
    float inv = 1.f / (sqrtf(nrm) + 1e-8f);
    float x2 = 0.f;
    #pragma unroll
    for (int c = 0; c < EMBD; ++c) { acc[c] *= inv; x2 += acc[c] * acc[c]; }

    float best = INFINITY; int bi = 0;
    for (int v = lane; v < NEMB; v += 32) {
        const float* e = emb + (size_t)v * EMBD;
        float dot = 0.f, e2 = 0.f;
        #pragma unroll
        for (int c = 0; c < EMBD; ++c) { float ev = e[c]; dot += acc[c] * ev; e2 += ev * ev; }
        float d = x2 - 2.f * dot + e2;
        if (d < best) { best = d; bi = v; }
    }
    for (int off = 16; off > 0; off >>= 1) {
        float ob = __shfl_xor(best, off); int oi = __shfl_xor(bi, off);
        if (ob < best || (ob == best && oi < bi)) { best = ob; bi = oi; }
    }
    if (lane == 0) {
        codes[row] = bi;
        smask[row] = (n * 4 + 3 < lens[b]) ? 1 : 0;
    }
}

// ---- fused GEMM (bf16 WMMA, async double-buffered LDS B) + online softmax ----
// grid (128 row-tiles of 64 rows, 4 N-slices of 2048 cols). 8 waves:
//   mi = wave&3 -> 16-row sub-tile, nj = wave>>2 -> 32-col half of each 64-col chunk.
__global__ __launch_bounds__(256) void k_main(const unsigned int* __restrict__ Apk,
                                              const unsigned int* __restrict__ Wp2,
                                              const int* __restrict__ codes,
                                              float* __restrict__ pMax, float* __restrict__ pSum,
                                              float* __restrict__ pTv,  float* __restrict__ pAmv,
                                              int*   __restrict__ pAmi) {
    __shared__ __attribute__((aligned(16))) unsigned int sA[64 * 256];       //  64 KB A tile
    __shared__ __attribute__((aligned(16))) unsigned int sB[2 * 16 * 1024];  // 2x64 KB B chunks
    __shared__ int   sTg[64];
    __shared__ float rM[8][16], rS[8][16], rT[8][16], rAv[8][16];
    __shared__ int   rAi[8][16];

    int tid = threadIdx.x;
    int rt  = blockIdx.x;                  // row tile (64 rows)
    int sl  = blockIdx.y;                  // N slice
    int b   = rt >> 3;
    int t0  = (rt & 7) << 6;
    int ntBase0 = sl * 128;                // 128 column-tiles (of 16) per slice

    unsigned sAoff  = lds_off(sA) + (unsigned)tid * 16u;
    unsigned sBoff0 = lds_off(sB) + (unsigned)tid * 16u;
    unsigned sBoff1 = sBoff0 + 65536u;

    {   // async DMA: A tile (64 rows x 512 K bf16 = 64 KB), 256 B per thread
        const unsigned char* g = (const unsigned char*)(Apk + (size_t)(b * TENC + t0) * 256)
                               + (size_t)tid * 16;
        #pragma unroll
        for (int i = 0; i < 16; ++i) async_cp16(sAoff + i * 4096u, g + (size_t)i * 4096);
    }
    {   // async DMA: first B chunk into buffer 0
        const unsigned char* g = (const unsigned char*)Wp2 + (size_t)ntBase0 * 16384
                               + (size_t)tid * 16;
        #pragma unroll
        for (int i = 0; i < 16; ++i) async_cp16(sBoff0 + i * 4096u, g + (size_t)i * 4096);
    }
    if (tid < 64) {
        int t = t0 + tid;
        sTg[tid] = (t <= 510) ? codes[b * TENC + t + 1] : -1;
    }

    int wave = tid >> 5, lane = tid & 31;
    int mi = wave & 3, nj = wave >> 2;
    int hlf = lane >> 4, lo = lane & 15;

    float rmax[8], rsum[8], amv[8], tv[8]; int ami[8];
    #pragma unroll
    for (int r = 0; r < 8; ++r) { rmax[r] = -INFINITY; rsum[r] = 0.f; amv[r] = -INFINITY; ami[r] = 0; tv[r] = -INFINITY; }

    for (int it = 0; it < 32; ++it) {
        wait_async0();                     // own async copies landed in LDS
        __syncthreads();                   // everyone's copies visible; prev readers done
        if (it + 1 < 32) {                 // prefetch next chunk into the other buffer
            const unsigned char* g = (const unsigned char*)Wp2
                                   + (size_t)(ntBase0 + (it + 1) * 4) * 16384
                                   + (size_t)tid * 16;
            unsigned dst = ((it + 1) & 1) ? sBoff1 : sBoff0;
            #pragma unroll
            for (int i = 0; i < 16; ++i) async_cp16(dst + i * 4096u, g + (size_t)i * 4096);
        }

        v8f c0 = {0,0,0,0,0,0,0,0}, c1 = {0,0,0,0,0,0,0,0};
        const uint4* a4 = (const uint4*)sA;
        const uint4* b4 = (const uint4*)(sB + (it & 1) * 16384);
        int ntl0 = nj * 2, ntl1 = ntl0 + 1;
        #pragma unroll 4
        for (int kk = 0; kk < 16; ++kk) {
            Frag af, bf0, bf1;
            int abase = (mi * 16 + lo) * 64 + kk * 4 + hlf * 2;   // uint4 units
            af.q[0] = a4[abase];
            af.q[1] = a4[abase + 1];
            int bb0 = ((ntl0 * 16 + kk) * 32 + lane) * 2;
            int bb1 = ((ntl1 * 16 + kk) * 32 + lane) * 2;
            bf0.q[0] = b4[bb0]; bf0.q[1] = b4[bb0 + 1];
            bf1.q[0] = b4[bb1]; bf1.q[1] = b4[bb1 + 1];
            c0 = __builtin_amdgcn_wmma_f32_16x16x32_bf16(false, af.v, false, bf0.v, (short)0, c0, false, false);
            c1 = __builtin_amdgcn_wmma_f32_16x16x32_bf16(false, af.v, false, bf1.v, (short)0, c1, false, false);
        }
        int colA = sl * 2048 + it * 64 + nj * 32 + lo, colB = colA + 16;
        #pragma unroll
        for (int r = 0; r < 8; ++r) {
            int m = mi * 16 + hlf * 8 + r;
            float v0 = c0[r], v1 = c1[r];
            int ti = sTg[m];
            if (colA == ti) tv[r] = v0;
            if (colB == ti) tv[r] = v1;
            if (v0 > amv[r]) { amv[r] = v0; ami[r] = colA; }
            if (v1 > amv[r]) { amv[r] = v1; ami[r] = colB; }
            float mn = fmaxf(rmax[r], fmaxf(v0, v1));
            rsum[r] = rsum[r] * __expf(rmax[r] - mn) + __expf(v0 - mn) + __expf(v1 - mn);
            rmax[r] = mn;
        }
    }

    // combine the 16 columns inside each half-wave (same 8-row set)
    #pragma unroll
    for (int r = 0; r < 8; ++r) {
        for (int off = 1; off < 16; off <<= 1) {
            float om = __shfl_xor(rmax[r], off), os = __shfl_xor(rsum[r], off);
            float M  = fmaxf(rmax[r], om);
            rsum[r]  = rsum[r] * __expf(rmax[r] - M) + os * __expf(om - M);
            rmax[r]  = M;
            float oav = __shfl_xor(amv[r], off); int oai = __shfl_xor(ami[r], off);
            if (oav > amv[r] || (oav == amv[r] && oai < ami[r])) { amv[r] = oav; ami[r] = oai; }
            tv[r] = fmaxf(tv[r], __shfl_xor(tv[r], off));
        }
    }
    if (lo == 0) {
        #pragma unroll
        for (int r = 0; r < 8; ++r) {
            int rr = hlf * 8 + r;              // row within 16-row sub-tile
            rM[wave][rr] = rmax[r]; rS[wave][rr] = rsum[r];
            rT[wave][rr] = tv[r];   rAv[wave][rr] = amv[r]; rAi[wave][rr] = ami[r];
        }
    }
    __syncthreads();

    if (tid < 64) {                            // merge nj=0 / nj=1 waves per row
        int m = tid, mi2 = m >> 4, rr = m & 15;
        int w0 = mi2, w1 = mi2 + 4;
        float m0 = rM[w0][rr], m1 = rM[w1][rr];
        float M  = fmaxf(m0, m1);
        float S  = rS[w0][rr] * __expf(m0 - M) + rS[w1][rr] * __expf(m1 - M);
        float T  = fmaxf(rT[w0][rr], rT[w1][rr]);
        float AV = rAv[w0][rr]; int AI = rAi[w0][rr];
        float av1 = rAv[w1][rr]; int ai1 = rAi[w1][rr];
        if (av1 > AV || (av1 == AV && ai1 < AI)) { AV = av1; AI = ai1; }
        size_t o = (size_t)sl * (BATCH * TENC) + rt * 64 + m;
        pMax[o] = M; pSum[o] = S; pTv[o] = T; pAmv[o] = AV; pAmi[o] = AI;
    }
}

// ---- per-row: merge 4 N-slice partials, masked NLL/accuracy/presence ----
__global__ void k_rows(const float* __restrict__ pMax, const float* __restrict__ pSum,
                       const float* __restrict__ pTv,  const float* __restrict__ pAmv,
                       const int*   __restrict__ pAmi,
                       const int* __restrict__ codes, const int* __restrict__ smask,
                       const int* __restrict__ lens,
                       int* __restrict__ presence, float* __restrict__ acc3) {
    int row = blockIdx.x * 256 + threadIdx.x;      // 8192 rows
    int b = row >> 9, t = row & 511;
    float M = -INFINITY, S = 0.f, T = -INFINITY, AV = -INFINITY; int AI = 0;
    for (int s = 0; s < NSLICE; ++s) {
        size_t o = (size_t)s * (BATCH * TENC) + row;
        float om = pMax[o], os = pSum[o];
        float M2 = fmaxf(M, om);
        S = S * __expf(M - M2) + os * __expf(om - M2);
        M = M2;
        float oav = pAmv[o]; int oai = pAmi[o];
        if (oav > AV || (oav == AV && oai < AI)) { AV = oav; AI = oai; }
        T = fmaxf(T, pTv[o]);
    }
    if (t <= 510) {
        int j  = t + 1;
        int mk = ((j < lens[b] / 4) ? 1 : 0) & smask[b * TENC + j];
        int ti = codes[b * TENC + j];
        if (mk) {
            float nll = M + __logf(S) - T;         // logsumexp - logit[target]
            atomicAdd(&acc3[0], nll);
            atomicAdd(&acc3[1], (AI == ti) ? 1.f : 0.f);
            atomicAdd(&acc3[2], 1.f);
        }
        presence[mk ? ti : 0] = 1;                 // masked_tgt scatter
    }
}

// ---------------- final reduce: uniq count + scalars -> d_out[4] ----------------
__global__ void k_final(const int* __restrict__ presence, const float* __restrict__ acc3,
                        float* __restrict__ out) {
    __shared__ int red[256];
    int tid = threadIdx.x, s = 0;
    for (int i = tid; i < NEMB; i += 256) s += presence[i];
    red[tid] = s;
    __syncthreads();
    for (int off = 128; off > 0; off >>= 1) { if (tid < off) red[tid] += red[tid + off]; __syncthreads(); }
    if (tid == 0) {
        float cnt = acc3[2];
        out[0] = acc3[0] / cnt;     // loss
        out[1] = acc3[1] / cnt;     // codes_acc
        out[2] = cnt;               // num_codes (NCB=1)
        out[3] = (float)red[0];     // uniq_num_codes
    }
}

extern "C" void kernel_launch(void* const* d_in, const int* in_sizes, int n_in,
                              void* d_out, int out_size, void* d_ws, size_t ws_size,
                              hipStream_t stream) {
    const float* feats = (const float*)d_in[0];
    const int*   lens  = (const int*)d_in[1];
    const float* enc   = (const float*)d_in[2];
    const float* proj  = (const float*)d_in[3];
    const float* emb   = (const float*)d_in[4];
    const float* top   = (const float*)d_in[5];
    float* out = (float*)d_out;

    char* ws = (char*)d_ws;
    unsigned int* Wp2 = (unsigned int*)(ws);                                // 8 MB
    unsigned int* Apk = (unsigned int*)(ws + (size_t)8 * 1024 * 1024);      // 8 MB
    int* codes    = (int*)(ws + (size_t)16 * 1024 * 1024);                  // 32 KB
    int* smaskp   = codes + NEMB;                                           // 32 KB
    int* presence = smaskp + NEMB;                                          // 32 KB
    float* acc3   = (float*)(presence + NEMB);                              // 64 B pad
    float* pMax   = acc3 + 16;                                              // 4*8192 each
    float* pSum   = pMax + NSLICE * BATCH * TENC;
    float* pTv    = pSum + NSLICE * BATCH * TENC;
    float* pAmv   = pTv  + NSLICE * BATCH * TENC;
    int*   pAmi   = (int*)(pAmv + NSLICE * BATCH * TENC);

    k_init <<<(NEMB + 255) / 256, 256, 0, stream>>>(presence, acc3);
    k_packW<<<8192, 256, 0, stream>>>(top, Wp2);
    k_packA<<<8192, 256, 0, stream>>>(enc, Apk);
    k_codes<<<1024, 256, 0, stream>>>(feats, lens, proj, emb, codes, smaskp);
    k_main <<<dim3(128, NSLICE), 256, 0, stream>>>(Apk, Wp2, codes, pMax, pSum, pTv, pAmv, pAmi);
    k_rows <<<BATCH * TENC / 256, 256, 0, stream>>>(pMax, pSum, pTv, pAmv, pAmi,
                                                    codes, smaskp, lens, presence, acc3);
    k_final<<<1, 256, 0, stream>>>(presence, acc3, out);

    (void)in_sizes; (void)n_in; (void)out_size; (void)ws_size;
}
